// ModulationLinear_86079734546951
// MI455X (gfx1250) — matrix operand order repaired and checked by reading, not compiled, not run on hardware
//
#include <hip/hip_runtime.h>

// ---------------------------------------------------------------------------
// ModulationLinear fused kernel for MI455X (gfx1250, wave32, WMMA).
//
// Roofline: main GEMM is 65536x512x512 (34.4 GFLOP bf16). HBM floor = 268 MB
// (input read + output write) = ~11.5 us @ 23.3 TB/s. bf16 WMMA compute is in
// the same ballpark, so both data movement AND weight-reuse matter:
//  - 64-row M-strip per block => normalized-weight L2 traffic = 1024 blocks
//    x 512 KB = 512 MB (vs 2 GB with 16-row strips).
//  - per wave: 4(M) x 2(N) accumulator tiles => each B fragment feeds 4 WMMAs,
//    each A fragment feeds 2; explicit double-buffering pipelines the
//    global/LDS fragment loads under the 8 WMMAs of the previous K-step.
// Both GEMM operands are L2-normalized rows (values ~1/sqrt(512)) => bf16 is
// well conditioned.
// ---------------------------------------------------------------------------

typedef __attribute__((ext_vector_type(16))) __bf16 v16bf;
typedef __attribute__((ext_vector_type(8)))  __bf16 v8bf;
typedef __attribute__((ext_vector_type(8)))  float  v8f;

union AV { v16bf v; v8bf h[2]; };

#define INV_SQRT_C 0.04419417382415922f   // 1/sqrt(512)
#define EPS        1e-8f
#define LRELU      0.2f
#define LGAIN      1.4142135623730951f    // sqrt(2)

// LDS geometry: 64 rows x 520 bf16 (1040 B stride -> rows shift 4 banks each,
// conflict-free ds_load_b128 across 16 rows), then reduction scratch.
#define XT_STRIDE  520
#define XT_BYTES   (64 * XT_STRIDE * 2)            // 66560
#define RED_BYTES  (32 * 16 * 4)                   // 2048
#define RN_BYTES   (32 * 4)                        // 128
#define SMEM_BYTES (XT_BYTES + RED_BYTES + RN_BYTES)

__device__ __forceinline__ __bf16 f2bf(float f) {
    union { float f; unsigned u; } a; a.f = f;
    unsigned u = a.u;
    u += 0x7FFFu + ((u >> 16) & 1u);
    union { unsigned short s; __bf16 b; } o;
    o.s = (unsigned short)(u >> 16);
    return o.b;
}

// --------------------------------------------------------------------------
// Kernel 1: gamma[b][c] = (1/sqrt(Z)) * dot(z[b,:], mod_weight[c,:]) + mod_bias[c]
// --------------------------------------------------------------------------
__global__ void gamma_kernel(const float* __restrict__ z,
                             const float* __restrict__ mw,
                             const float* __restrict__ mb,
                             float* __restrict__ gamma) {
    int gid = blockIdx.x * blockDim.x + threadIdx.x;   // 0..4095
    int b = gid >> 9;
    int c = gid & 511;
    const float* zr = z  + b * 512;
    const float* wr = mw + c * 512;
    float acc = 0.f;
#pragma unroll 8
    for (int k = 0; k < 512; ++k) acc = fmaf(zr[k], wr[k], acc);
    gamma[gid] = acc * INV_SQRT_C + mb[c];
}

// --------------------------------------------------------------------------
// Kernel 2: wn[o][c] = bf16( (weight[o][c]/sqrt(C)) * rsqrt(sum_c w^2 + eps) )
// --------------------------------------------------------------------------
__global__ void wnorm_kernel(const float* __restrict__ weight,
                             __bf16* __restrict__ wn) {
    int o = blockIdx.x;
    int t = threadIdx.x;
    float a0 = weight[o * 512 + t]       * INV_SQRT_C;
    float a1 = weight[o * 512 + t + 256] * INV_SQRT_C;

    __shared__ float red[256];
    red[t] = a0 * a0 + a1 * a1;
    __syncthreads();
    for (int off = 128; off > 0; off >>= 1) {
        if (t < off) red[t] += red[t + off];
        __syncthreads();
    }
    __shared__ float rn_s;
    if (t == 0) rn_s = rsqrtf(red[0] + EPS);
    __syncthreads();
    float r = rn_s;
    wn[o * 512 + t]       = f2bf(a0 * r);
    wn[o * 512 + t + 256] = f2bf(a1 * r);
}

// --------------------------------------------------------------------------
// Kernel 3: fused modulate + row-normalize + bf16 WMMA GEMM + bias/leakyrelu.
// Block = 64 M-rows x full 512 N. 512 threads = 16 waves; wave wv owns
// columns [wv*32, wv*32+32) and all 4 M sub-tiles -> 8 accumulators.
// K = 512 -> 16 double-buffered steps of 8x v_wmma_f32_16x16x32_bf16.
// --------------------------------------------------------------------------
__global__ __launch_bounds__(512, 1) void fused_kernel(
        const float*  __restrict__ input,
        const float*  __restrict__ gamma,
        const __bf16* __restrict__ wn,
        const float*  __restrict__ act_bias,
        float*        __restrict__ out) {
    extern __shared__ char smem[];
    __bf16* xtile = (__bf16*)smem;                       // [64][520]
    float*  red   = (float*)(smem + XT_BYTES);           // [32][16]
    float*  rn    = (float*)(smem + XT_BYTES + RED_BYTES); // [32]

    const int  tid = threadIdx.x;
    const long m0  = (long)blockIdx.x * 64;

    // ---------------- Phase 1: modulate + normalize 64 rows into LDS -------
    {
        const int r16  = tid >> 4;          // 0..31 (row within pass)
        const int q    = tid & 15;          // 16 threads per row
        const int col0 = q * 32;            // 32 columns per thread
#pragma unroll 1
        for (int p = 0; p < 2; ++p) {
            const int  r    = p * 32 + r16;          // row in strip 0..63
            const long grow = m0 + r;                // global row
            const int  b    = (int)(grow >> 13);     // batch (8192 rows each)

            const float4* in4 = (const float4*)(input + grow * 512 + col0);
            const float4* gm4 = (const float4*)(gamma + (long)b * 512 + col0);

            float vals[32];
            float ss = 0.f;
#pragma unroll
            for (int jj = 0; jj < 8; ++jj) {
                float4 iv = in4[jj];
                float4 gv = gm4[jj];
                float x0 = iv.x * gv.x, x1 = iv.y * gv.y;
                float x2 = iv.z * gv.z, x3 = iv.w * gv.w;
                vals[jj * 4 + 0] = x0; vals[jj * 4 + 1] = x1;
                vals[jj * 4 + 2] = x2; vals[jj * 4 + 3] = x3;
                ss += x0 * x0 + x1 * x1 + x2 * x2 + x3 * x3;
            }
            red[r16 * 16 + q] = ss;
            __syncthreads();
            if (q == 0) {
                float s = 0.f;
#pragma unroll
                for (int k = 0; k < 16; ++k) s += red[r16 * 16 + k];
                rn[r16] = rsqrtf(s + EPS);
            }
            __syncthreads();
            float sc = rn[r16];
#pragma unroll
            for (int jj = 0; jj < 4; ++jj) {
                v8bf pk;
#pragma unroll
                for (int j = 0; j < 8; ++j) pk[j] = f2bf(vals[jj * 8 + j] * sc);
                *(v8bf*)&xtile[r * XT_STRIDE + col0 + jj * 8] = pk;
            }
            __syncthreads();   // protect red/rn for next pass & xtile for GEMM
        }
    }

    // ---------------- Phase 2: WMMA GEMM over K=512 ------------------------
    const int wv   = tid >> 5;        // wave 0..15 -> columns [wv*32, +32)
    const int lane = tid & 31;
    const int h    = lane >> 4;       // half-wave select
    const int m    = lane & 15;       // M (A) / N (B,C,D) lane index
    const int n0   = wv * 32;

    // A 16x32 tile, this code's k order: lane holds K = 16h..16h+15 contiguous (same as B)
    const __bf16* ap[4];
#pragma unroll
    for (int mt = 0; mt < 4; ++mt) ap[mt] = xtile + (mt * 16 + m) * XT_STRIDE + 16 * h;
    // B 32x16 tile, this code's k order: lane holds column n, K = 16h..16h+15 contiguous (same as A)
    const __bf16* wp[2];
    wp[0] = wn + (long)(n0 +  0 + m) * 512 + 16 * h;
    wp[1] = wn + (long)(n0 + 16 + m) * 512 + 16 * h;

    v8f acc[4][2] = {};

    // double-buffered fragments
    AV    a_c[4];
    v16bf b_c[2];
#pragma unroll
    for (int mt = 0; mt < 4; ++mt) {
        a_c[mt].h[0] = *(const v8bf*)(ap[mt]);
        a_c[mt].h[1] = *(const v8bf*)(ap[mt] + 8);
    }
    b_c[0] = *(const v16bf*)(wp[0]);
    b_c[1] = *(const v16bf*)(wp[1]);

#pragma unroll
    for (int kb = 0; kb < 16; ++kb) {
        AV    a_n[4];
        v16bf b_n[2];
        if (kb < 15) {
            const int o = (kb + 1) * 32;
#pragma unroll
            for (int mt = 0; mt < 4; ++mt) {
                a_n[mt].h[0] = *(const v8bf*)(ap[mt] + o);
                a_n[mt].h[1] = *(const v8bf*)(ap[mt] + o + 8);
            }
            b_n[0] = *(const v16bf*)(wp[0] + o);
            b_n[1] = *(const v16bf*)(wp[1] + o);
        }
#pragma unroll
        for (int mt = 0; mt < 4; ++mt) {
            acc[mt][0] = __builtin_amdgcn_wmma_f32_16x16x32_bf16(
                false, a_c[mt].v, false, b_c[0], (short)0, acc[mt][0], false, false);
            acc[mt][1] = __builtin_amdgcn_wmma_f32_16x16x32_bf16(
                false, a_c[mt].v, false, b_c[1], (short)0, acc[mt][1], false, false);
        }
        if (kb < 15) {
#pragma unroll
            for (int mt = 0; mt < 4; ++mt) a_c[mt] = a_n[mt];
            b_c[0] = b_n[0];
            b_c[1] = b_n[1];
        }
    }

    // ---------------- Epilogue: bias + leaky relu * sqrt(2) ----------------
    // C/D layout: VGPR rr holds (M = rr + 8h, N = m).
#pragma unroll
    for (int nt = 0; nt < 2; ++nt) {
        const int   n    = n0 + nt * 16 + m;
        const float bias = act_bias[n];
#pragma unroll
        for (int mt = 0; mt < 4; ++mt) {
#pragma unroll
            for (int rr = 0; rr < 8; ++rr) {
                float y = acc[mt][nt][rr] + bias;
                y = (y > 0.f ? y : LRELU * y) * LGAIN;
                out[(m0 + mt * 16 + rr + 8 * h) * 512 + n] = y;
            }
        }
    }
}

// --------------------------------------------------------------------------
extern "C" void kernel_launch(void* const* d_in, const int* in_sizes, int n_in,
                              void* d_out, int out_size, void* d_ws, size_t ws_size,
                              hipStream_t stream) {
    const float* input      = (const float*)d_in[0];  // [8, 8192, 512]
    const float* z          = (const float*)d_in[1];  // [8, 512]
    const float* weight     = (const float*)d_in[2];  // [512, 512]
    const float* mod_weight = (const float*)d_in[3];  // [512, 512]
    const float* mod_bias   = (const float*)d_in[4];  // [512]
    const float* act_bias   = (const float*)d_in[5];  // [512]
    float* out = (float*)d_out;                        // [8, 8192, 512]

    // workspace: gamma f32 [8*512] (16 KB) | wn bf16 [512*512] (512 KB)
    float*  gamma = (float*)d_ws;
    __bf16* wn    = (__bf16*)((char*)d_ws + 16384);

    gamma_kernel<<<16,   256, 0, stream>>>(z, mod_weight, mod_bias, gamma);
    wnorm_kernel<<<512,  256, 0, stream>>>(weight, wn);
    fused_kernel<<<1024, 512, SMEM_BYTES, stream>>>(input, gamma, wn, act_bias, out);
}
